// VectorQuantizer_21036749816007
// MI455X (gfx1250) — matrix operand order repaired and checked by reading, not compiled, not run on hardware
//
#include <hip/hip_runtime.h>
#include <hip/hip_bf16.h>

// ---------------------------------------------------------------------------
// VQ-VAE vector quantizer for MI455X (gfx1250, wave32, WMMA bf16 16x16x32)
//
// N = 131072 rows, C = 64, K = 512 codes.
// score(n,k) = ||e_k||^2 - 2 * x_n . e_k   (||x||^2 dropped: constant per row)
// loss = 1.25 * mean((emb[argmin] - x)^2)
//
// CDNA5 paths: v_wmma_f32_16x16x32_bf16 (2 independent accumulator chains),
// global_load_async_to_lds_b128 + s_wait_asynccnt for codebook staging,
// ds_load_b128 fragment feeds, ds_bpermute argmin reduction.
// ---------------------------------------------------------------------------

typedef __attribute__((ext_vector_type(16))) __bf16        v16bf;
typedef __attribute__((ext_vector_type(8)))  float         v8f;
typedef __attribute__((ext_vector_type(8)))  unsigned short us8;

union FragAB {
    v16bf bf;
    us8   u[2];   // two 16-byte halves -> ds_load_b128 each
};

#define NUM_CODES   512
#define EMB_DIM     64
#define SPATIAL     32768            // 32*32*32
#define BATCH_CH    (EMB_DIM * SPATIAL)
#define N_ROWS      131072           // 4 * 32768
#define ROWS_PER_BLK 128
#define PAD_STRIDE  72               // ushorts per padded row (144 B, 16B aligned)
#define QUANT_ELEMS 8388608          // 4*64*32768
#define LOSS_POS    QUANT_ELEMS
#define IDX_POS     (QUANT_ELEMS + 1)

// round-to-nearest-even f32 -> bf16 bits
static __device__ __forceinline__ unsigned short f32_to_bf16(float f) {
    unsigned int u = __float_as_uint(f);
    unsigned int r = u + 0x7FFFu + ((u >> 16) & 1u);
    return (unsigned short)(r >> 16);
}

static __device__ __forceinline__ v8f wmma_bf16(v16bf a, v16bf b, v8f c) {
    return __builtin_amdgcn_wmma_f32_16x16x32_bf16(
        false, a, false, b, (short)0, c, false, false);
}

// ---------------------------------------------------------------------------
// Kernel A: codebook f32 -> bf16 (ws[0..65535]) and ||e_k||^2 (ws+65536)
// ---------------------------------------------------------------------------
__global__ void vq_prep_kernel(const float* __restrict__ emb,
                               unsigned short* __restrict__ emb_bf,
                               float* __restrict__ norms) {
    int row = blockIdx.x * blockDim.x + threadIdx.x;
    if (row >= NUM_CODES) return;
    float ss = 0.0f;
    #pragma unroll 8
    for (int c = 0; c < EMB_DIM; ++c) {
        float v = emb[row * EMB_DIM + c];
        ss += v * v;
        emb_bf[row * EMB_DIM + c] = f32_to_bf16(v);
    }
    norms[row] = ss;
}

// ---------------------------------------------------------------------------
// Kernel B: zero the loss accumulator (atomics target)
// ---------------------------------------------------------------------------
__global__ void vq_zero_loss_kernel(float* __restrict__ out) {
    if (threadIdx.x == 0) out[LOSS_POS] = 0.0f;
}

// ---------------------------------------------------------------------------
// Kernel C: fused GEMM + argmin + gather + loss.
// Grid: N_ROWS / ROWS_PER_BLK = 1024 blocks, 256 threads (8 wave32).
// ---------------------------------------------------------------------------
__global__ void __launch_bounds__(256)
vq_main_kernel(const float* __restrict__ in,          // [4,64,32,32,32]
               const float* __restrict__ emb_f32,     // [512,64]
               const unsigned short* __restrict__ emb_bf,  // ws codebook bf16
               const float* __restrict__ norms_g,     // ws norms
               float* __restrict__ out) {             // quantized | loss | indices
    __shared__ __attribute__((aligned(16))) unsigned short emb_lds[NUM_CODES * PAD_STRIDE];
    __shared__ __attribute__((aligned(16))) unsigned short x_lds[ROWS_PER_BLK * PAD_STRIDE];
    __shared__ float norms_lds[NUM_CODES];
    __shared__ int   idx_lds[ROWS_PER_BLK];
    __shared__ float red_lds[8];

    const int tid  = threadIdx.x;
    const int wave = tid >> 5;
    const int lane = tid & 31;
    const int rowBase = blockIdx.x * ROWS_PER_BLK;
    const int b  = rowBase >> 15;        // / 32768
    const int s0 = rowBase & 32767;

    // ---- stage codebook bf16 into padded LDS via async copy (ASYNCcnt) ----
    // 64 KB = 4096 x 16B chunks; per-lane VDST carries the padded LDS addr.
    for (int i = tid; i < NUM_CODES * (EMB_DIM / 8); i += 256) {
        int row   = i >> 3;              // 8 x 16B chunks per code row
        int off16 = i & 7;
        unsigned lds_b = (unsigned)(unsigned long long)
            (&emb_lds[row * PAD_STRIDE + off16 * 8]);
        int goff = i * 16;
        asm volatile("global_load_async_to_lds_b128 %0, %1, %2"
                     :: "v"(lds_b), "v"(goff), "s"(emb_bf)
                     : "memory");
    }
    for (int i = tid; i < NUM_CODES; i += 256) norms_lds[i] = norms_g[i];

    // ---- stage x tile: read coalesced along spatial, store transposed -----
    // input element (b, c, s0+sloc) -> x_lds[sloc][c] as bf16 (pairs of c)
    for (int j = 0; j < 16; ++j) {
        int e    = j * 256 + tid;        // 4096 c-pairs
        int cp   = e >> 7;               // c pair 0..31
        int sloc = e & 127;
        size_t base = (size_t)b * BATCH_CH + (size_t)(2 * cp) * SPATIAL + (size_t)(s0 + sloc);
        float x0 = in[base];
        float x1 = in[base + SPATIAL];
        unsigned int packed = (unsigned int)f32_to_bf16(x0) |
                              ((unsigned int)f32_to_bf16(x1) << 16);
        *(unsigned int*)&x_lds[sloc * PAD_STRIDE + 2 * cp] = packed;
    }
    asm volatile("s_wait_asynccnt 0x0" ::: "memory");
    __syncthreads();

    // ---- per-wave GEMM + argmin over the 16 rows this wave owns -----------
    const int r = lane & 15;             // column-in-tile / row-in-A
    const int h = lane >> 4;             // lane half

    // A fragments (reused across all 32 code tiles): contiguous k order chosen by this code
    // lane half h holds K = kc*32 + 16h + e, contiguous 16 bf16 (matches B)
    FragAB afr[2];
    const unsigned short* xrow = &x_lds[(wave * 16 + r) * PAD_STRIDE];
    #pragma unroll
    for (int kc = 0; kc < 2; ++kc) {
        afr[kc].u[0] = *(const us8*)&xrow[kc * 32 + 16 * h];
        afr[kc].u[1] = *(const us8*)&xrow[kc * 32 + 16 * h + 8];
    }

    float best[8];
    int   bidx[8];
    #pragma unroll
    for (int v = 0; v < 8; ++v) { best[v] = 3.4e38f; bidx[v] = 0; }

    // two code tiles per iteration -> two independent WMMA accumulator
    // chains, hiding the bf16 WMMA result-hazard window behind issue of the
    // sibling tile's WMMAs and the previous tiles' compare VALU work.
    for (int t = 0; t < 32; t += 2) {
        const int code0 = t * 16 + r;    // this lane's code column, tile t
        const int code1 = code0 + 16;    // tile t+1
        // B layout: lane half h holds K = kc*32 + 16h + e, contiguous 16 bf16
        const unsigned short* e0 = &emb_lds[code0 * PAD_STRIDE + 16 * h];
        const unsigned short* e1 = &emb_lds[code1 * PAD_STRIDE + 16 * h];
        FragAB b00, b01, b10, b11;
        b00.u[0] = *(const us8*)&e0[0];   b00.u[1] = *(const us8*)&e0[8];
        b01.u[0] = *(const us8*)&e0[32];  b01.u[1] = *(const us8*)&e0[40];
        b10.u[0] = *(const us8*)&e1[0];   b10.u[1] = *(const us8*)&e1[8];
        b11.u[0] = *(const us8*)&e1[32];  b11.u[1] = *(const us8*)&e1[40];

        v8f acc0 = {}, acc1 = {};
        acc0 = wmma_bf16(afr[0].bf, b00.bf, acc0);
        acc1 = wmma_bf16(afr[0].bf, b10.bf, acc1);
        acc0 = wmma_bf16(afr[1].bf, b01.bf, acc0);
        acc1 = wmma_bf16(afr[1].bf, b11.bf, acc1);

        const float n0 = norms_lds[code0];
        const float n1 = norms_lds[code1];
        #pragma unroll
        for (int v = 0; v < 8; ++v) {    // row M = v + 8h, col N = r
            float s0v = n0 - 2.0f * acc0[v];
            if (s0v < best[v]) { best[v] = s0v; bidx[v] = code0; }
            float s1v = n1 - 2.0f * acc1[v];
            if (s1v < best[v]) { best[v] = s1v; bidx[v] = code1; }
        }
    }

    // cross-lane argmin within each 16-lane half (rows live per-half)
    #pragma unroll
    for (int v = 0; v < 8; ++v) {
        float bv = best[v];
        int   bi = bidx[v];
        #pragma unroll
        for (int m = 1; m < 16; m <<= 1) {
            float ov = __shfl_xor(bv, m, 32);
            int   oi = __shfl_xor(bi, m, 32);
            if (ov < bv || (ov == bv && oi < bi)) { bv = ov; bi = oi; }
        }
        if (r == 0) idx_lds[wave * 16 + v + 8 * h] = bi;
    }
    __syncthreads();

    // ---- phase 2: gather f32 code rows -> output, fused loss --------------
    float lsum = 0.0f;
    for (int j = 0; j < 32; ++j) {
        int e    = j * 256 + tid;        // 8192 elements
        int c    = e >> 7;
        int sloc = e & 127;
        int code = idx_lds[sloc];
        float qv = emb_f32[code * EMB_DIM + c];
        size_t off = (size_t)b * BATCH_CH + (size_t)c * SPATIAL + (size_t)(s0 + sloc);
        float xv = in[off];
        out[off] = qv;
        float d = qv - xv;
        lsum += d * d;
    }
    #pragma unroll
    for (int m = 16; m >= 1; m >>= 1) lsum += __shfl_xor(lsum, m, 32);
    if (lane == 0) red_lds[wave] = lsum;
    __syncthreads();
    if (tid == 0) {
        float s = 0.0f;
        #pragma unroll
        for (int w = 0; w < 8; ++w) s += red_lds[w];
        // loss = (1 + 0.25) * sum / (N * C)
        atomicAdd(&out[LOSS_POS], s * (1.25f / (float)QUANT_ELEMS));
    }

    // ---- indices output (as float values, output buffer is f32) -----------
    if (tid < ROWS_PER_BLK) {
        out[IDX_POS + rowBase + tid] = (float)idx_lds[tid];
    }
}

// ---------------------------------------------------------------------------
extern "C" void kernel_launch(void* const* d_in, const int* in_sizes, int n_in,
                              void* d_out, int out_size, void* d_ws, size_t ws_size,
                              hipStream_t stream) {
    const float* in      = (const float*)d_in[0];   // [4,64,32,32,32]
    const float* emb_f32 = (const float*)d_in[1];   // [512,64]
    float* out = (float*)d_out;

    unsigned short* emb_bf = (unsigned short*)d_ws;                       // 64 KB
    float* norms = (float*)((char*)d_ws + NUM_CODES * EMB_DIM * 2);       // 2 KB

    vq_prep_kernel<<<2, 256, 0, stream>>>(emb_f32, emb_bf, norms);
    vq_zero_loss_kernel<<<1, 32, 0, stream>>>(out);
    vq_main_kernel<<<N_ROWS / ROWS_PER_BLK, 256, 0, stream>>>(
        in, emb_f32, emb_bf, norms, out);
}
